// Basic_Block_v1_53369263620389
// MI455X (gfx1250) — matrix operand order repaired and checked by reading, not compiled, not run on hardware
//
#include <hip/hip_runtime.h>
#include <math.h>

// ---------------------------------------------------------------------------
// MI455X (gfx1250, wave32) implementation of the Mamba2-vision block.
// All dense contractions run on v_wmma_f32_16x16x32_f16 (f32 accumulate),
// 2x2 C-tiles per wave (4 independent WMMAs per k-step). Weights and all
// GEMM-feeding activations are stored once as f16; the sequential SSD scan
// keeps its 64x64 state in registers (16 f32/lane).
// ---------------------------------------------------------------------------

typedef __attribute__((ext_vector_type(16))) _Float16 v16h;
typedef __attribute__((ext_vector_type(8)))  _Float16 v8h;
typedef __attribute__((ext_vector_type(8)))  float    v8f;

#define LN_EPS 1e-5f

// ---------------- block reductions (power-of-2 blockDim only) --------------
__device__ __forceinline__ float block_sum(float v, float* red) {
  int t = threadIdx.x;
  red[t] = v; __syncthreads();
  for (int s = blockDim.x >> 1; s > 0; s >>= 1) {
    if (t < s) red[t] += red[t + s];
    __syncthreads();
  }
  float r = red[0]; __syncthreads();
  return r;
}
__device__ __forceinline__ float block_max(float v, float* red) {
  int t = threadIdx.x;
  red[t] = v; __syncthreads();
  for (int s = blockDim.x >> 1; s > 0; s >>= 1) {
    if (t < s) red[t] = fmaxf(red[t], red[t + s]);
    __syncthreads();
  }
  float r = red[0]; __syncthreads();
  return r;
}

// ---------------- f32 -> f16 conversion (weights, staged once) -------------
__global__ void cvt_h_k(const float* __restrict__ in, _Float16* __restrict__ out, long n) {
  long i = (long)blockIdx.x * blockDim.x + threadIdx.x;
  if (i < n) out[i] = (_Float16)in[i];
}

// ---------------------------------------------------------------------------
// WMMA GEMM, f16 inputs, f32 accumulate. One wave -> 32x32 C block
// (2x2 tiles of 16x16, 4 accumulators, 4 independent WMMAs per k-step).
//   BT=true : C[m][n] = sum_k A[m][k] * B[n][k]   (B row-major [N][K])
//   BT=false: C[m][n] = sum_k A[m][k] * B[k][n]   (B row-major [K][N])
// Optional bias[n], residual add, f16 copy of C. Batched via strides.
// blockDim = 128 (4 waves -> 128 rows per block). K%32==0, M%32==0, N guarded.
// ---------------------------------------------------------------------------
template<bool BT, bool BIAS, bool RES>
__global__ void gemm_wmma(const _Float16* __restrict__ A, const _Float16* __restrict__ B,
                          const float* __restrict__ bias, const float* __restrict__ res,
                          float* __restrict__ C, _Float16* __restrict__ Ch,
                          int M, int N, int K, long asb, long bsb, long csb) {
  int batch = blockIdx.z;
  A += (long)batch * asb;
  B += (long)batch * bsb;
  C += (long)batch * csb;
  const float* resp = RES ? (res + (long)batch * csb) : nullptr;
  _Float16* chp = Ch ? (Ch + (long)batch * csb) : nullptr;

  int wave = threadIdx.x >> 5;
  int lane = threadIdx.x & 31;
  int tn = blockIdx.x * 32;
  int tm = (blockIdx.y * 4 + wave) * 32;
  if (tm >= M) return;                      // wave-uniform: EXEC stays all-1s

  int r      = lane & 15;
  int hi     = (lane >= 16);
  int koff8  = hi ? 16 : 0;                 // A fragment K halves
  int koff16 = hi ? 16 : 0;                 // B fragment K halves
  int n0 = tn + r, n1 = tn + 16 + r;
  bool ok0 = (n0 < N), ok1 = (n1 < N);

  const _Float16* A0r = A + (long)(tm + r) * K + koff8;
  const _Float16* A1r = A0r + (long)16 * K;

  auto loadB = [&](int n, bool ok, int k0) -> v16h {
    v8h lo = {}, hi8 = {};
    if (ok) {
      if (BT) {
        const _Float16* Bp = B + (long)n * K + k0 + koff16;
        lo  = *(const v8h*)Bp;
        hi8 = *(const v8h*)(Bp + 8);
      } else {
        const _Float16* Bp = B + (long)(k0 + koff16) * N + n;
#pragma unroll
        for (int e = 0; e < 8; ++e) {
          lo[e]  = Bp[(long)e * N];
          hi8[e] = Bp[(long)(e + 8) * N];
        }
      }
    }
    return __builtin_shufflevector(lo, hi8, 0, 1, 2, 3, 4, 5, 6, 7,
                                   8, 9, 10, 11, 12, 13, 14, 15);
  };

  v8f acc00 = {}, acc01 = {}, acc10 = {}, acc11 = {};
  for (int k0 = 0; k0 < K; k0 += 32) {
    // A fragments (16B vector loads; contiguous K halves k = 16h + s, matching B)
    v8h a0lo = *(const v8h*)(A0r + k0);
    v8h a0hi = *(const v8h*)(A0r + k0 + 8);
    v8h a1lo = *(const v8h*)(A1r + k0);
    v8h a1hi = *(const v8h*)(A1r + k0 + 8);
    v16h a0 = __builtin_shufflevector(a0lo, a0hi, 0, 1, 2, 3, 4, 5, 6, 7,
                                      8, 9, 10, 11, 12, 13, 14, 15);
    v16h a1 = __builtin_shufflevector(a1lo, a1hi, 0, 1, 2, 3, 4, 5, 6, 7,
                                      8, 9, 10, 11, 12, 13, 14, 15);
    v16h b0 = loadB(n0, ok0, k0);
    v16h b1 = loadB(n1, ok1, k0);

    __builtin_prefetch((const void*)(A0r + k0 + 32), 0, 1);   // global_prefetch
    __builtin_prefetch((const void*)(A1r + k0 + 32), 0, 1);

    acc00 = __builtin_amdgcn_wmma_f32_16x16x32_f16(false, a0, false, b0,
                                                   (short)0, acc00, false, false);
    acc01 = __builtin_amdgcn_wmma_f32_16x16x32_f16(false, a0, false, b1,
                                                   (short)0, acc01, false, false);
    acc10 = __builtin_amdgcn_wmma_f32_16x16x32_f16(false, a1, false, b0,
                                                   (short)0, acc10, false, false);
    acc11 = __builtin_amdgcn_wmma_f32_16x16x32_f16(false, a1, false, b1,
                                                   (short)0, acc11, false, false);
  }

  auto store = [&](const v8f& acc, int rb, int cb) {
    int col = cb + r;
    if (col >= N) return;
    float bv = BIAS ? bias[col] : 0.f;
    int rbase = rb + (hi ? 8 : 0);
#pragma unroll
    for (int j = 0; j < 8; ++j) {
      long idx = (long)(rbase + j) * N + col;
      float vv = acc[j] + bv;
      if (RES) vv += resp[idx];
      C[idx] = vv;
      if (chp) chp[idx] = (_Float16)vv;
    }
  };
  store(acc00, tm, tn);
  store(acc01, tm, tn + 16);
  store(acc10, tm + 16, tn);
  store(acc11, tm + 16, tn + 16);
}

// ---------------- sin/cos 2D embedding + sorted gather ---------------------
__global__ void embed_sort_k(const float* __restrict__ x, const int* __restrict__ sidx,
                             float* __restrict__ xs) {
  int b = blockIdx.x, l = blockIdx.y, c = threadIdx.x;   // c in [0,128)
  int p  = sidx[b * 256 + l];
  int hh = p >> 4, ww = p & 15;
  int half = c >> 6;
  int cc   = c & 63;
  float pos = half ? (float)ww : (float)hh;
  int j = cc & 31;
  float omega = expf(-(float)j * (9.210340371976184f / 32.f)); // 10000^(-j/32)
  float t = pos * omega;
  float pe = (cc < 32) ? sinf(t) : cosf(t);
  xs[((long)b * 256 + l) * 128 + c] =
      x[(((long)b * 128 + c) * 16 + hh) * 16 + ww] + pe;
}

// ---------------- LayerNorm over last dim (blockDim == C) ------------------
__global__ void layernorm_k(const float* __restrict__ x, const float* __restrict__ w,
                            const float* __restrict__ bb, float* __restrict__ out,
                            _Float16* __restrict__ outh, int C) {
  __shared__ float red[256];
  long row = blockIdx.x;
  int t = threadIdx.x;
  float v = x[row * C + t];
  float mu = block_sum(v, red) / (float)C;
  float d = v - mu;
  float var = block_sum(d * d, red) / (float)C;
  float o = d * rsqrtf(var + LN_EPS) * w[t] + bb[t];
  out[row * C + t] = o;
  if (outh) outh[row * C + t] = (_Float16)o;
}

// ---------------- causal depthwise conv (K=4) + silu, softplus(dt) ---------
__global__ void convdt_k(const float* __restrict__ zx, const float* __restrict__ cw,
                         const float* __restrict__ cb, const float* __restrict__ dtb,
                         float* __restrict__ xbc, float* __restrict__ dt,
                         int L, int dproj, int dinner, int dconv, int nheads) {
  int b = blockIdx.x, l = blockIdx.y, ch = threadIdx.x;   // ch in [0,dconv)
  long row = (long)b * L + l;
  float acc = cb[ch];
#pragma unroll
  for (int k = 0; k < 4; ++k) {
    int ls = l - 3 + k;
    if (ls >= 0)
      acc += zx[((long)b * L + ls) * dproj + dinner + ch] * cw[ch * 4 + k];
  }
  xbc[row * dconv + ch] = acc / (1.f + expf(-acc));        // silu
  if (ch < nheads) {
    float dv = zx[row * dproj + dinner + dconv + ch] + dtb[ch];
    dt[row * nheads + ch] = (dv > 20.f) ? dv : log1pf(expf(dv));  // softplus
  }
}

// ---------------- SSD scan: one block per (batch, head), state in regs -----
__global__ void ssd_scan_k(const float* __restrict__ xbc, const float* __restrict__ dt,
                           const float* __restrict__ A_log, float* __restrict__ y,
                           int L, int dinner, int dconv, int nheads) {
  int bh = blockIdx.x;
  int b = bh / nheads, h = bh % nheads;
  __shared__ float sX[64], sB[64], sC[64];
  __shared__ float sdt;
  int t = threadIdx.x;
  int p = t >> 2, nb = (t & 3) << 4;
  float A = -expf(A_log[h]);
  float hs[16];
#pragma unroll
  for (int j = 0; j < 16; ++j) hs[j] = 0.f;

  for (int l = 0; l < L; ++l) {
    long row = (long)b * L + l;
    if (t < 64)       sX[t]       = xbc[row * dconv + h * 64 + t];
    else if (t < 128) sB[t - 64]  = xbc[row * dconv + dinner + (t - 64)];
    else if (t < 192) sC[t - 128] = xbc[row * dconv + dinner + 64 + (t - 128)];
    else if (t == 192) sdt = dt[row * nheads + h];
    __syncthreads();
    float dtv = sdt;
    float dA  = expf(dtv * A);
    float dtx = dtv * sX[p];
    float part = 0.f;
#pragma unroll
    for (int j = 0; j < 16; ++j) {
      hs[j] = hs[j] * dA + dtx * sB[nb + j];
      part += hs[j] * sC[nb + j];
    }
    part += __shfl_xor(part, 1, 32);
    part += __shfl_xor(part, 2, 32);
    if ((t & 3) == 0) y[row * dinner + h * 64 + p] = part;
    __syncthreads();
  }
}

// ---------------- (y + D*xh) * silu(z), RMSNorm * rms_w -> f16 -------------
__global__ void gate_rms_k(const float* __restrict__ y, const float* __restrict__ xbc,
                           const float* __restrict__ zx, const float* __restrict__ D,
                           const float* __restrict__ rmsw, _Float16* __restrict__ out,
                           int dproj, int dinner, int dconv) {
  __shared__ float red[512];
  long row = blockIdx.x;
  int t = threadIdx.x;
  float yv = y[row * dinner + t] + D[t >> 6] * xbc[row * dconv + t];
  float z  = zx[row * dproj + t];
  float val = yv * (z / (1.f + expf(-z)));
  float ms = block_sum(val * val, red) / (float)dinner;
  out[row * dinner + t] = (_Float16)(val * rsqrtf(ms + LN_EPS) * rmsw[t]);
}

// ---------------- transpose (b,I,J) -> (b,J,I), opt residual, opt f16 ------
__global__ void transpose_k(const float* __restrict__ in, const float* __restrict__ res,
                            float* __restrict__ out, _Float16* __restrict__ outh,
                            int I, int J, long total) {
  long idx = (long)blockIdx.x * blockDim.x + threadIdx.x;
  if (idx >= total) return;
  int j  = (int)(idx % J);
  long r = idx / J;
  int i  = (int)(r % I);
  int bb = (int)(r / I);
  long o = ((long)bb * J + j) * I + i;
  float v = in[((long)bb * I + i) * J + j];
  if (res) v += res[o];
  out[o] = v;
  if (outh) outh[o] = (_Float16)v;
}

// ---------------- center projection: einsum('blc,ocl->bo'), l<5 ------------
__global__ void center_k(const float* __restrict__ xs, const float* __restrict__ w,
                         const float* __restrict__ bias, float* __restrict__ center) {
  int b = blockIdx.x, o = threadIdx.x;
  float acc = bias[o];
  for (int l = 0; l < 5; ++l)
    for (int c = 0; c < 128; ++c)
      acc += xs[((long)b * 256 + l) * 128 + c] * w[(o * 128 + c) * 5 + l];
  center[b * 128 + o] = acc;
}

// ---------------- q = center @ aq_w.T + aq_b -------------------------------
__global__ void qproj_k(const float* __restrict__ center, const float* __restrict__ w,
                        const float* __restrict__ bias, float* __restrict__ q) {
  int b = blockIdx.x, o = threadIdx.x;
  float acc = bias[o];
  for (int c = 0; c < 128; ++c) acc += center[b * 128 + c] * w[o * 128 + c];
  q[b * 128 + o] = acc;
}

// ---------------- attention-1: softmax_l(q.k / 4), hd=16, nh=8 -------------
__global__ void attn1_k(const float* __restrict__ q, const float* __restrict__ k,
                        float* __restrict__ attn) {
  __shared__ float red[256];
  int b = blockIdx.x, h = blockIdx.y, l = threadIdx.x;
  float logit = 0.f;
#pragma unroll
  for (int d = 0; d < 16; ++d)
    logit += q[b * 128 + h * 16 + d] * k[((long)b * 256 + l) * 128 + h * 16 + d];
  logit *= 0.25f;
  float mx = block_max(logit, red);
  float e = expf(logit - mx);
  float s = block_sum(e, red);
  attn[((long)b * 8 + h) * 256 + l] = e / s;
}

// ---------------- vw[b,l,c] = v[b,l,c] * attn[b, c>>4, l]  (f16 out) -------
__global__ void vscale_k(const float* __restrict__ v, const float* __restrict__ attn,
                         _Float16* __restrict__ vw, long total) {
  long idx = (long)blockIdx.x * blockDim.x + threadIdx.x;
  if (idx >= total) return;
  int c  = (int)(idx % 128);
  long r = idx / 128;
  int l  = (int)(r % 256);
  int b  = (int)(r / 256);
  vw[idx] = (_Float16)(v[idx] * attn[((long)b * 8 + (c >> 4)) * 256 + l]);
}

// ---------------- in-place row softmax with scale + f16 copy ---------------
__global__ void softmax_row_k(float* __restrict__ a, _Float16* __restrict__ ah,
                              int Ncol, float scale) {
  __shared__ float red[256];
  long row = blockIdx.x;
  int t = threadIdx.x;
  float v = a[row * Ncol + t] * scale;
  float mx = block_max(v, red);
  float e = expf(v - mx);
  float s = block_sum(e, red);
  float o = e / s;
  a[row * Ncol + t] = o;
  if (ah) ah[row * Ncol + t] = (_Float16)o;
}

// ---------------- unsort scatter: xr[b, sidx[b,l], c] = xs[b,l,c] ----------
__global__ void scatter_k(const float* __restrict__ xs, const int* __restrict__ sidx,
                          float* __restrict__ xr, long total) {
  long idx = (long)blockIdx.x * blockDim.x + threadIdx.x;
  if (idx >= total) return;
  int c  = (int)(idx % 128);
  long r = idx / 128;
  int l  = (int)(r % 256);
  int b  = (int)(r / 256);
  int p  = sidx[b * 256 + l];
  xr[((long)b * 256 + p) * 128 + c] = xs[idx];
}

// ---------------- im2col (3x3 s2 p1 on (b,16,16,128)) -> f16 ---------------
__global__ void im2col_k(const float* __restrict__ xr, _Float16* __restrict__ col) {
  long idx = (long)blockIdx.x * blockDim.x + threadIdx.x;
  if (idx >= (long)1024 * 1152) return;
  int cidx = (int)(idx % 1152);
  long r   = idx / 1152;
  int pix  = (int)(r & 63);
  int b    = (int)(r >> 6);
  int oh = pix >> 3, ow = pix & 7;
  int kh = cidx / 384;
  int rem = cidx % 384;
  int kw = rem / 128, ci = rem % 128;
  int ih = oh * 2 - 1 + kh, iw = ow * 2 - 1 + kw;
  float v = 0.f;
  if (ih >= 0 && ih < 16 && iw >= 0 && iw < 16)
    v = xr[(((long)b * 16 + ih) * 16 + iw) * 128 + ci];
  col[idx] = (_Float16)v;
}

// ---------------------------------------------------------------------------
// host-side GEMM dispatcher
// ---------------------------------------------------------------------------
static void gemm(hipStream_t s, const _Float16* A, const _Float16* B, const float* bias,
                 const float* res, float* C, _Float16* Ch, int M, int N, int K,
                 int batch = 1, long asb = 0, long bsb = 0, long csb = 0,
                 bool bt = true) {
  dim3 g((N + 31) / 32, (M + 127) / 128, batch), blk(128, 1, 1);
  if (bt) {
    if (bias && res)
      gemm_wmma<true, true, true><<<g, blk, 0, s>>>(A, B, bias, res, C, Ch, M, N, K, asb, bsb, csb);
    else if (bias)
      gemm_wmma<true, true, false><<<g, blk, 0, s>>>(A, B, bias, res, C, Ch, M, N, K, asb, bsb, csb);
    else if (res)
      gemm_wmma<true, false, true><<<g, blk, 0, s>>>(A, B, bias, res, C, Ch, M, N, K, asb, bsb, csb);
    else
      gemm_wmma<true, false, false><<<g, blk, 0, s>>>(A, B, bias, res, C, Ch, M, N, K, asb, bsb, csb);
  } else {
    gemm_wmma<false, false, false><<<g, blk, 0, s>>>(A, B, bias, res, C, Ch, M, N, K, asb, bsb, csb);
  }
}

extern "C" void kernel_launch(void* const* d_in, const int* in_sizes, int n_in,
                              void* d_out, int out_size, void* d_ws, size_t ws_size,
                              hipStream_t stream) {
  (void)in_sizes; (void)n_in; (void)out_size; (void)ws_size;

  const float* x         = (const float*)d_in[0];
  const int*   sidx      = (const int*)  d_in[1];
  const float* spa_ln_w  = (const float*)d_in[2];
  const float* spa_ln_b  = (const float*)d_in[3];
  const float* spa_in_w  = (const float*)d_in[4];
  const float* spa_cw    = (const float*)d_in[5];
  const float* spa_cb    = (const float*)d_in[6];
  const float* spa_dtb   = (const float*)d_in[7];
  const float* spa_Alog  = (const float*)d_in[8];
  const float* spa_D     = (const float*)d_in[9];
  const float* spa_rmsw  = (const float*)d_in[10];
  const float* spa_out_w = (const float*)d_in[11];
  const float* spe_ln_w  = (const float*)d_in[12];
  const float* spe_ln_b  = (const float*)d_in[13];
  const float* spe_in_w  = (const float*)d_in[14];
  const float* spe_cw    = (const float*)d_in[15];
  const float* spe_cb    = (const float*)d_in[16];
  const float* spe_dtb   = (const float*)d_in[17];
  const float* spe_Alog  = (const float*)d_in[18];
  const float* spe_D     = (const float*)d_in[19];
  const float* spe_rmsw  = (const float*)d_in[20];
  const float* spe_out_w = (const float*)d_in[21];
  const float* norm_w    = (const float*)d_in[22];
  const float* norm_b    = (const float*)d_in[23];
  const float* cprj_w    = (const float*)d_in[24];
  const float* cprj_b    = (const float*)d_in[25];
  const float* aq_w = (const float*)d_in[26]; const float* aq_b = (const float*)d_in[27];
  const float* ak_w = (const float*)d_in[28]; const float* ak_b = (const float*)d_in[29];
  const float* av_w = (const float*)d_in[30]; const float* av_b = (const float*)d_in[31];
  const float* ao_w = (const float*)d_in[32]; const float* ao_b = (const float*)d_in[33];
  const float* sq_w = (const float*)d_in[34]; const float* sq_b = (const float*)d_in[35];
  const float* sk_w = (const float*)d_in[36]; const float* sk_b = (const float*)d_in[37];
  const float* sv_w = (const float*)d_in[38]; const float* sv_b = (const float*)d_in[39];
  const float* so_w = (const float*)d_in[40]; const float* so_b = (const float*)d_in[41];
  const float* ds_cw   = (const float*)d_in[42];
  const float* ds_ln_w = (const float*)d_in[43];
  const float* ds_ln_b = (const float*)d_in[44];

  // ---------------- f32 workspace arena ----------------
  float* W = (float*)d_ws;
  size_t off = 0;
  auto alloc = [&](size_t n) { float* p = W + off; off += n; return p; };
  float* XS  = alloc(524288);   // (16,256,128) running activations
  float* H1  = alloc(524288);   // spatial-mamba residual output
  float* XT  = alloc(524288);   // transposed (16,128,256) workspace
  float* U   = alloc(524288);   // LN outputs / xr
  float* T0  = alloc(2637824);  // zxbcdt (max 16*256*644)
  float* XBC = alloc(1572864);  // conv output (max 16*256*384)
  float* DTb = alloc(32768);    // dt
  float* Y   = alloc(1048576);  // scan output / q2 / conv out
  float* YN  = alloc(1048576);  // y2 f32
  float* Kb  = alloc(524288);   // k / k2
  float* Vb  = alloc(524288);   // v / v2
  float* ATT = alloc(262144);   // attn (16,8,256) / attn2 (16,128,128)
  float* SM  = alloc(4096);     // center (2048) + q (2048)

  // ---------------- f16 workspace arena (16B-aligned) ----------------
  _Float16* HB = (_Float16*)(W + off);
  size_t ho = 0;
  auto halloc = [&](size_t n) { _Float16* p = HB + ho; ho += n; return p; };
  _Float16* WH_spa_in  = halloc(2 * 644 * 128);
  _Float16* WH_spa_out = halloc(2 * 128 * 256);
  _Float16* WH_spe_in  = halloc(2 * 1160 * 256);
  _Float16* WH_spe_out = halloc(2 * 256 * 512);
  _Float16* WH_ak = halloc(16384);
  _Float16* WH_av = halloc(16384);
  _Float16* WH_ao = halloc(16384);
  _Float16* WH_sq = halloc(65536);
  _Float16* WH_sk = halloc(65536);
  _Float16* WH_sv = halloc(65536);
  _Float16* WH_so = halloc(65536);
  _Float16* WH_ds = halloc(147456);
  _Float16* UH   = halloc(524288);
  _Float16* YNH  = halloc(1048576);  // gate / vw / y2 (f16 GEMM inputs)
  _Float16* XTH  = halloc(524288);
  _Float16* QH   = halloc(524288);
  _Float16* KH   = halloc(524288);
  _Float16* VH   = halloc(524288);
  _Float16* ATTH = halloc(262144);
  _Float16* COLH = halloc(1179648);

  auto cvt = [&](const float* src, _Float16* dst, long n) {
    cvt_h_k<<<(n + 255) / 256, 256, 0, stream>>>(src, dst, n);
  };

  // stage weights as f16 once per call (deterministic; no caching allowed)
  cvt(spa_in_w,  WH_spa_in,  2L * 644 * 128);
  cvt(spa_out_w, WH_spa_out, 2L * 128 * 256);
  cvt(spe_in_w,  WH_spe_in,  2L * 1160 * 256);
  cvt(spe_out_w, WH_spe_out, 2L * 256 * 512);
  cvt(ak_w, WH_ak, 16384); cvt(av_w, WH_av, 16384); cvt(ao_w, WH_ao, 16384);
  cvt(sq_w, WH_sq, 65536); cvt(sk_w, WH_sk, 65536);
  cvt(sv_w, WH_sv, 65536); cvt(so_w, WH_so, 65536);
  cvt(ds_cw, WH_ds, 147456);

  const long TOT = 524288;      // 16*256*128

  // 1) embedding + sorted gather
  embed_sort_k<<<dim3(16, 256), 128, 0, stream>>>(x, sidx, XS);

  // 2) two layers of (spatial mamba -> spectral mamba)
  for (int i = 0; i < 2; ++i) {
    // ---- spatial mamba: d_model=128, L=256, d_inner=256, nheads=4 ----
    {
      int rows = 16 * 256;
      layernorm_k<<<rows, 128, 0, stream>>>(XS, spa_ln_w + i * 128, spa_ln_b + i * 128,
                                            U, UH, 128);
      gemm(stream, UH, WH_spa_in + (long)i * 644 * 128, nullptr, nullptr, T0, nullptr,
           rows, 644, 128);
      convdt_k<<<dim3(16, 256), 384, 0, stream>>>(T0, spa_cw + i * 1536, spa_cb + i * 384,
                                                  spa_dtb + i * 4, XBC, DTb, 256, 644, 256, 384, 4);
      ssd_scan_k<<<16 * 4, 256, 0, stream>>>(XBC, DTb, spa_Alog + i * 4, Y, 256, 256, 384, 4);
      gate_rms_k<<<rows, 256, 0, stream>>>(Y, XBC, T0, spa_D + i * 4, spa_rmsw + i * 256, YNH,
                                           644, 256, 384);
      gemm(stream, YNH, WH_spa_out + (long)i * 128 * 256, nullptr, XS, H1, nullptr,
           rows, 128, 256);
    }
    // ---- spectral mamba: d_model=256, L=128, d_inner=512, nheads=8 ----
    {
      transpose_k<<<(TOT + 255) / 256, 256, 0, stream>>>(H1, nullptr, XT, nullptr, 256, 128, TOT);
      int rows = 16 * 128;
      layernorm_k<<<rows, 256, 0, stream>>>(XT, spe_ln_w + i * 256, spe_ln_b + i * 256,
                                            U, UH, 256);
      gemm(stream, UH, WH_spe_in + (long)i * 1160 * 256, nullptr, nullptr, T0, nullptr,
           rows, 1160, 256);
      convdt_k<<<dim3(16, 128), 640, 0, stream>>>(T0, spe_cw + i * 2560, spe_cb + i * 640,
                                                  spe_dtb + i * 8, XBC, DTb, 128, 1160, 512, 640, 8);
      ssd_scan_k<<<16 * 8, 256, 0, stream>>>(XBC, DTb, spe_Alog + i * 8, Y, 128, 512, 640, 8);
      gate_rms_k<<<rows, 512, 0, stream>>>(Y, XBC, T0, spe_D + i * 8, spe_rmsw + i * 512, YNH,
                                           1160, 512, 640);
      gemm(stream, YNH, WH_spe_out + (long)i * 256 * 512, nullptr, nullptr, XT, nullptr,
           rows, 256, 512);
      transpose_k<<<(TOT + 255) / 256, 256, 0, stream>>>(XT, H1, XS, nullptr, 128, 256, TOT);
    }
  }

  // 3) final LN over channels
  layernorm_k<<<16 * 256, 128, 0, stream>>>(XS, norm_w, norm_b, U, UH, 128);

  // 4) attention-1 (single query per head from center projection)
  center_k<<<16, 128, 0, stream>>>(U, cprj_w, cprj_b, SM);
  qproj_k<<<16, 128, 0, stream>>>(SM, aq_w, aq_b, SM + 2048);
  gemm(stream, UH, WH_ak, ak_b, nullptr, Kb, nullptr, 4096, 128, 128);
  gemm(stream, UH, WH_av, av_b, nullptr, Vb, nullptr, 4096, 128, 128);
  attn1_k<<<dim3(16, 8), 256, 0, stream>>>(SM + 2048, Kb, ATT);
  vscale_k<<<(TOT + 255) / 256, 256, 0, stream>>>(Vb, ATT, YNH, TOT);
  gemm(stream, YNH, WH_ao, ao_b, U, XS, nullptr, 4096, 128, 128);  // xs = LN(xs)+out

  // 5) attention-2 over transposed features (b,128,256)
  transpose_k<<<(TOT + 255) / 256, 256, 0, stream>>>(XS, nullptr, XT, XTH, 256, 128, TOT);
  gemm(stream, XTH, WH_sq, sq_b, nullptr, Y,  QH, 2048, 256, 256);
  gemm(stream, XTH, WH_sk, sk_b, nullptr, Kb, KH, 2048, 256, 256);
  gemm(stream, XTH, WH_sv, sv_b, nullptr, Vb, VH, 2048, 256, 256);
  // logits[b] = q2 @ k2^T  (batched, BT=true)
  gemm(stream, QH, KH, nullptr, nullptr, ATT, nullptr, 128, 128, 256,
       16, 128L * 256, 128L * 256, 128L * 128, true);
  softmax_row_k<<<2048, 128, 0, stream>>>(ATT, ATTH, 128, 1.f / 16.f);
  // y2[b] = attn2 @ v2  (batched, BT=false)
  gemm(stream, ATTH, VH, nullptr, nullptr, YN, YNH, 128, 256, 128,
       16, 128L * 128, 128L * 256, 128L * 256, false);
  gemm(stream, YNH, WH_so, so_b, nullptr, XT, nullptr, 2048, 256, 256);
  transpose_k<<<(TOT + 255) / 256, 256, 0, stream>>>(XT, XS, XS, nullptr, 128, 256, TOT);

  // 6) unsort, 3x3/s2 conv via f16 im2col + WMMA GEMM, final LN -> d_out
  scatter_k<<<(TOT + 255) / 256, 256, 0, stream>>>(XS, sidx, U, TOT);
  im2col_k<<<((long)1024 * 1152 + 255) / 256, 256, 0, stream>>>(U, COLH);
  gemm(stream, COLH, WH_ds, nullptr, nullptr, Y, nullptr, 1024, 128, 1152,
       1, 0, 0, 0, false);
  layernorm_k<<<1024, 128, 0, stream>>>(Y, ds_ln_w, ds_ln_b, (float*)d_out, nullptr, 128);
}
